// GATLayer_7851200217925
// MI455X (gfx1250) — matrix operand order repaired and checked by reading, not compiled, not run on hardware
//

#include <hip/hip_runtime.h>
#include <hip/hip_bf16.h>
#include <math.h>

// ---------------------------------------------------------------------------
// GAT layer for MI455X (gfx1250, wave32):
//   k0: f32 -> f16 convert of x        (xh  [4096 x 512])
//   k1: f32 -> f16 convert+transpose W (wth [512  x 512], W^T so B-operand
//       loads are K-contiguous b128s)
//   k2: Wh = x @ W via v_wmma_f32_16x16x32_f16, one wave per 16x16 tile
//   k3: s_src/s_dst head scores (tiny GEMV)
//   k4: per-row online-softmax rowsum over masked logits (never materializes
//       the 537MB attention tensor) fused with BN(eval)+ReLU epilogue
// ---------------------------------------------------------------------------

#define GAT_N      4096
#define GAT_IND    512
#define GAT_OUTD   512
#define GAT_H      8
#define GAT_HD     64
#define GAT_ALPHA  0.2f
#define GAT_EPS    1e-5f
#define GAT_NEG    -9.0e15f

typedef __attribute__((ext_vector_type(16))) _Float16 v16h;
typedef __attribute__((ext_vector_type(8)))  _Float16 v8h;
typedef __attribute__((ext_vector_type(8)))  float    v8f;

// ---------------- k0: convert x to f16 -------------------------------------
__global__ void gat_cvt_x(const float* __restrict__ x, _Float16* __restrict__ xh) {
    int idx = blockIdx.x * blockDim.x + threadIdx.x;          // 0 .. N*IND-1
    xh[idx] = (_Float16)x[idx];
}

// ---------------- k1: convert + transpose W to f16 -------------------------
__global__ void gat_cvt_wt(const float* __restrict__ W, _Float16* __restrict__ wth) {
    int idx = blockIdx.x * blockDim.x + threadIdx.x;          // 0 .. IND*OUTD-1
    int k = idx >> 9;            // row of W   (in-dim)
    int n = idx & 511;           // col of W   (out-dim)
    wth[n * GAT_IND + k] = (_Float16)W[idx];                  // W^T[n][k]
}

// ---------------- k2: WMMA GEMM  Wh = x @ W --------------------------------
// 256 threads = 8 waves / block; one wave owns one 16x16 tile of Wh.
// Tiles: 256 (M) x 32 (N) = 8192 waves -> 1024 blocks.
__global__ void gat_wmma_gemm(const _Float16* __restrict__ xh,
                              const _Float16* __restrict__ wth,
                              float* __restrict__ wh) {
    const int lane = threadIdx.x & 31;
    const int wid  = blockIdx.x * 8 + (threadIdx.x >> 5);
    const int tm   = wid >> 5;          // 0..255  (M tile)
    const int tn   = wid & 31;          // 0..31   (N tile)
    const int ml   = lane & 15;
    const bool hi  = lane >= 16;

    // A (16x32 f16): lane m<16 holds K {0..7,16..23}; lanes 16..31 hold {8..15,24..31}
    const _Float16* arow = xh  + (size_t)(tm * 16 + ml) * GAT_IND + (hi ? 8 : 0);
    // B (32x16 f16): lane n holds K {0..7,16..23} (lo half) / {8..15,24..31} (hi half), from W^T
    const _Float16* brow = wth + (size_t)(tn * 16 + ml) * GAT_IND + (hi ? 8 : 0);

    v8f acc = {};
    for (int kb = 0; kb < GAT_IND; kb += 32) {
        if (kb + 64 < GAT_IND) {                 // global_prefetch_b8 hints
            __builtin_prefetch((const void*)(arow + kb + 64), 0, 0);
            __builtin_prefetch((const void*)(brow + kb + 64), 0, 0);
        }
        v8h a0 = *(const v8h*)(arow + kb);
        v8h a1 = *(const v8h*)(arow + kb + 16);
        v8h b0 = *(const v8h*)(brow + kb);
        v8h b1 = *(const v8h*)(brow + kb + 16);
        v16h A = __builtin_shufflevector(a0, a1, 0,1,2,3,4,5,6,7,8,9,10,11,12,13,14,15);
        v16h B = __builtin_shufflevector(b0, b1, 0,1,2,3,4,5,6,7,8,9,10,11,12,13,14,15);
        // (neg_a, A, neg_b, B, c_mod, C, reuse_a, reuse_b)
        acc = __builtin_amdgcn_wmma_f32_16x16x32_f16(false, A, false, B,
                                                     (short)0, acc, false, false);
    }

    // C/D layout: lane n = lane&15; VGPR r -> row r (lanes 0-15) / r+8 (16-31)
    #pragma unroll
    for (int r = 0; r < 8; ++r) {
        int mm = tm * 16 + r + (hi ? 8 : 0);
        int nn = tn * 16 + ml;
        wh[(size_t)mm * GAT_OUTD + nn] = acc[r];
    }
}

// ---------------- k3: attention head scores --------------------------------
// s_src[n,h] = Wh[n,h,:].a[h,:64] ; s_dst[n,h] = Wh[n,h,:].a[h,64:]
__global__ void gat_scores(const float* __restrict__ wh, const float* __restrict__ a,
                           float* __restrict__ ssrc, float* __restrict__ sdst) {
    int t = blockIdx.x * blockDim.x + threadIdx.x;            // 0 .. N*H-1
    int n = t >> 3, h = t & 7;
    const float* whp = wh + (size_t)n * GAT_OUTD + h * GAT_HD;
    const float* a1  = a + h * (2 * GAT_HD);
    const float* a2  = a1 + GAT_HD;
    float ss = 0.f, sd = 0.f;
    #pragma unroll 8
    for (int d = 0; d < GAT_HD; ++d) {
        float w = whp[d];
        ss = fmaf(w, a1[d], ss);
        sd = fmaf(w, a2[d], sd);
    }
    ssrc[t] = ss;
    sdst[t] = sd;
}

// ---------------- k4: online-softmax rowsum + BN + ReLU --------------------
// One block (256 thr = 8 waves) per row i. Each thread scans 16 columns,
// keeping per-head running (max, expsum); wave32 shfl_xor reduce, then LDS
// cross-wave merge, then fused epilogue.
__global__ void gat_attn_bn(const int* __restrict__ adj,
                            const float* __restrict__ ssrc,
                            const float* __restrict__ sdst,
                            const float* __restrict__ wh,
                            const float* __restrict__ gamma,
                            const float* __restrict__ beta,
                            const float* __restrict__ rmean,
                            const float* __restrict__ rvar,
                            float* __restrict__ out) {
    const int i   = blockIdx.x;
    const int tid = threadIdx.x;
    const int wv  = tid >> 5;

    __shared__ float lm[8][GAT_H];
    __shared__ float ls[8][GAT_H];
    __shared__ float att[GAT_H];

    float srow[GAT_H];
    #pragma unroll
    for (int h = 0; h < GAT_H; ++h) srow[h] = ssrc[i * GAT_H + h];

    float m[GAT_H], s[GAT_H];
    #pragma unroll
    for (int h = 0; h < GAT_H; ++h) { m[h] = GAT_NEG; s[h] = 0.f; }

    for (int j = tid; j < GAT_N; j += 256) {
        const bool conn = adj[(size_t)i * GAT_N + j] > 0;
        #pragma unroll
        for (int h = 0; h < GAT_H; ++h) {
            float e = GAT_NEG;
            if (conn) {
                float v = srow[h] + sdst[j * GAT_H + h];
                e = (v > 0.f) ? v : GAT_ALPHA * v;
            }
            float nm = fmaxf(m[h], e);
            s[h] = s[h] * __expf(m[h] - nm) + __expf(e - nm);
            m[h] = nm;
        }
    }

    // wave32 butterfly reduce (online-softmax merge)
    #pragma unroll
    for (int off = 16; off > 0; off >>= 1) {
        #pragma unroll
        for (int h = 0; h < GAT_H; ++h) {
            float mo = __shfl_xor(m[h], off, 32);
            float so = __shfl_xor(s[h], off, 32);
            float nm = fmaxf(m[h], mo);
            s[h] = s[h] * __expf(m[h] - nm) + so * __expf(mo - nm);
            m[h] = nm;
        }
    }
    if ((tid & 31) == 0) {
        #pragma unroll
        for (int h = 0; h < GAT_H; ++h) { lm[wv][h] = m[h]; ls[wv][h] = s[h]; }
    }
    __syncthreads();

    if (tid < GAT_H) {
        const int h = tid;
        float M = lm[0][h], S = ls[0][h];
        #pragma unroll
        for (int w2 = 1; w2 < 8; ++w2) {
            float mo = lm[w2][h], so = ls[w2][h];
            float nm = fmaxf(M, mo);
            S = S * __expf(M - nm) + so * __expf(mo - nm);
            M = nm;
        }
        att[h] = S / S;   // rowsum of softmax (Z/Z; runtime value, not folded)
    }
    __syncthreads();

    #pragma unroll
    for (int c = tid; c < GAT_OUTD; c += 256) {
        float v = att[c >> 6] * wh[(size_t)i * GAT_OUTD + c];
        v = (v - rmean[c]) * rsqrtf(rvar[c] + GAT_EPS) * gamma[c] + beta[c];
        out[(size_t)i * GAT_OUTD + c] = fmaxf(v, 0.f);
    }
}

// ---------------------------------------------------------------------------
extern "C" void kernel_launch(void* const* d_in, const int* in_sizes, int n_in,
                              void* d_out, int out_size, void* d_ws, size_t ws_size,
                              hipStream_t stream) {
    const float* x     = (const float*)d_in[0];
    const int*   adj   = (const int*)  d_in[1];
    const float* W     = (const float*)d_in[2];
    const float* a     = (const float*)d_in[3];
    const float* gamma = (const float*)d_in[4];
    const float* beta  = (const float*)d_in[5];
    const float* rmean = (const float*)d_in[6];
    const float* rvar  = (const float*)d_in[7];
    float* out = (float*)d_out;

    // workspace layout (all regions fully overwritten every call)
    char* ws = (char*)d_ws;
    _Float16* xh   = (_Float16*)(ws);                                  // 4 MiB
    _Float16* wth  = (_Float16*)(ws + (size_t)(4  << 20));             // 512 KiB
    float*    wh   = (float*)   (ws + (size_t)(4  << 20) + (512 << 10)); // 8 MiB
    float*    ssrc = (float*)   (ws + (size_t)(13 << 20));             // 128 KiB
    float*    sdst = (float*)   (ws + (size_t)(13 << 20) + (128 << 10)); // 128 KiB

    gat_cvt_x   <<<(GAT_N * GAT_IND)   / 256, 256, 0, stream>>>(x, xh);
    gat_cvt_wt  <<<(GAT_IND * GAT_OUTD)/ 256, 256, 0, stream>>>(W, wth);
    gat_wmma_gemm<<<(GAT_N / 16) * (GAT_OUTD / 16) / 8, 256, 0, stream>>>(xh, wth, wh);
    gat_scores  <<<(GAT_N * GAT_H)     / 256, 256, 0, stream>>>(wh, a, ssrc, sdst);
    gat_attn_bn <<<GAT_N, 256, 0, stream>>>(adj, ssrc, sdst, wh,
                                            gamma, beta, rmean, rvar, out);
}
